// GNN_42056319762722
// MI455X (gfx1250) — matrix operand order repaired and checked by reading, not compiled, not run on hardware
//
#include <hip/hip_runtime.h>
#include <hip/hip_bf16.h>

#define N_NODES 50000
#define DEG     16
#define FDIM    128
#define HDIM    128
#define OUTD    64
#define G4      512          // 4*H gate rows
#define LDW     136          // padded LDS row stride (halves): 272B -> rotating banks
#define TM      16           // nodes per tile
#define NTILES  (N_NODES / TM)   // 3125

typedef __attribute__((ext_vector_type(16))) _Float16 v16h;
typedef __attribute__((ext_vector_type(8)))  _Float16 h8v;
typedef __attribute__((ext_vector_type(4)))  _Float16 h4v;
typedef __attribute__((ext_vector_type(8)))  float    v8f;

#if defined(__HIP_DEVICE_COMPILE__) && __has_builtin(__builtin_amdgcn_global_load_async_to_lds_b128)
#define HAVE_ASYNC_LDS 1
// builtin signature: (v4i AS1* src, v4i AS3* dst, imm int offset, imm int cpol)
typedef int v4i __attribute__((vector_size(16)));
typedef __attribute__((address_space(1))) v4i gv4i;
typedef __attribute__((address_space(3))) v4i lv4i;
#endif

// ---- WMMA fragment loaders (wave32, 16-bit operands in contiguous K order, same for A and B) ----
// A (16x32, MxK): lane l -> row M=l&15; khi=l>>4:
//   halves j -> K = k0 + khi*16 + j
__device__ __forceinline__ v16h load_A_frag(const _Float16* base, int k0, int lane) {
  int m   = lane & 15;
  int khi = lane >> 4;
  const _Float16* p = base + m * LDW + k0 + khi * 16;
  h8v lo = *(const h8v*)(p);        // 16B aligned
  h8v hi = *(const h8v*)(p + 8);
  v16h a;
#pragma unroll
  for (int j = 0; j < 8; ++j) { a[j] = lo[j]; a[j + 8] = hi[j]; }
  return a;
}

// B (32x16, KxN) for y = x @ W^T: B[k][n] = W[n0+n][k]
// lane l -> col N=l&15; khi=l>>4; halves j -> K = k0 + khi*16 + j
__device__ __forceinline__ v16h load_B_frag(const _Float16* base, int n0, int k0, int lane) {
  int n   = n0 + (lane & 15);
  int khi = lane >> 4;
  const _Float16* p = base + n * LDW + k0 + khi * 16;
  h8v lo = *(const h8v*)(p);
  h8v hi = *(const h8v*)(p + 8);
  v16h b;
#pragma unroll
  for (int j = 0; j < 8; ++j) { b[j] = lo[j]; b[j + 8] = hi[j]; }
  return b;
}

// Same B layout, loaded straight from the fp32 global weight matrix (row-major [rows][128]),
// converted to f16. Used once per kernel to make Whh fragments register-resident.
__device__ __forceinline__ v16h load_B_frag_global(const float* __restrict__ W, int n0, int k0, int lane) {
  int n   = n0 + (lane & 15);
  int khi = lane >> 4;
  const float4* p = (const float4*)(W + (size_t)n * FDIM + k0 + khi * 16);
  float4 a = p[0], b = p[1], c = p[2], d = p[3];
  v16h f;
  f[0]=(_Float16)a.x;  f[1]=(_Float16)a.y;  f[2]=(_Float16)a.z;  f[3]=(_Float16)a.w;
  f[4]=(_Float16)b.x;  f[5]=(_Float16)b.y;  f[6]=(_Float16)b.z;  f[7]=(_Float16)b.w;
  f[8]=(_Float16)c.x;  f[9]=(_Float16)c.y;  f[10]=(_Float16)c.z; f[11]=(_Float16)c.w;
  f[12]=(_Float16)d.x; f[13]=(_Float16)d.y; f[14]=(_Float16)d.z; f[15]=(_Float16)d.w;
  return f;
}

__device__ __forceinline__ v8f wmma_f16(v16h a, v16h b, v8f c) {
  return __builtin_amdgcn_wmma_f32_16x16x32_f16(false, a, false, b, (short)0, c, false, false);
}

__device__ __forceinline__ float fsigmoid(float x) { return 1.0f / (1.0f + __expf(-x)); }
__device__ __forceinline__ float ftanh(float x) {
  x = fminf(fmaxf(x, -15.0f), 15.0f);
  float e = __expf(2.0f * x);
  return (e - 1.0f) / (e + 1.0f);
}

// One SAGEConv(lstm) layer for a 16-node tile. 256 threads = 8 wave32.
// Wave w owns gate tiles {w, w+8, w+16, w+24} == i/f/g/o for hidden cols [16w,16w+16):
// LSTM pointwise + c-state fully in registers; Whh fragments register-resident; Wih in LDS;
// x_t gather double-buffered (async global->LDS on the f16 layer).
// TI = float (layer 1, fp32 node features) or _Float16 (layer 2, f16 activations).
template <typename TI>
__global__ void sage_lstm_layer(const TI*    __restrict__ xin,
                                const int*   __restrict__ src,
                                const float* __restrict__ wih,
                                const float* __restrict__ whh,
                                const float* __restrict__ bih,
                                const float* __restrict__ bhh,
                                const float* __restrict__ wl,
                                const float* __restrict__ bl,
                                const float* __restrict__ wr,
                                _Float16*    __restrict__ out)
{
  extern __shared__ _Float16 smem[];
  _Float16* sWih = smem;                 // [512][LDW] f16 (reused at end: lin_l rows 0..127, lin_r rows 128..255)
  _Float16* sXT0 = sWih + G4 * LDW;      // [16][LDW] gathered x_t (ping)
  _Float16* sXT1 = sXT0 + TM * LDW;      // [16][LDW] gathered x_t (pong)
  _Float16* sH   = sXT1 + TM * LDW;      // [16][LDW] hidden state
  _Float16* sXN  = sH   + TM * LDW;      // [16][LDW] node-own features

  const int tid  = threadIdx.x;          // 0..255
  const int lane = tid & 31;
  const int wave = tid >> 5;             // 0..7
  const int n0   = blockIdx.x * TM;

  // ---- Whh B-fragments: register-resident for all 16 steps (4 gates x 4 K-chunks) ----
  v16h whhF[4][4];
#pragma unroll
  for (int g = 0; g < 4; ++g)
#pragma unroll
    for (int kc = 0; kc < 4; ++kc)
      whhF[g][kc] = load_B_frag_global(whh, g * HDIM + wave * 16, kc * 32, lane);

  // ---- stage Wih fp32 -> f16 into LDS (L2-resident, shared by all 3125 WGs) ----
  {
    const float4* a4 = (const float4*)wih;
    for (int idx = tid; idx < G4 * (FDIM / 4); idx += 256) {
      int r = idx >> 5;                  // 32 float4 per row
      int c = (idx & 31) * 4;
      float4 va = a4[idx];
      h4v h; h[0]=(_Float16)va.x; h[1]=(_Float16)va.y; h[2]=(_Float16)va.z; h[3]=(_Float16)va.w;
      *(h4v*)(sWih + r * LDW + c) = h;
    }
  }
  // node-own features + zero h0 + stage x_t for t=0
  for (int idx = tid; idx < TM * FDIM; idx += 256) {
    int m = idx >> 7, c = idx & 127;
    sXN[m * LDW + c] = (_Float16)xin[(size_t)(n0 + m) * FDIM + c];
    sH [m * LDW + c] = (_Float16)0.0f;
    int sr0 = src[(n0 + m) * DEG + 0];
    sXT0[m * LDW + c] = (_Float16)xin[(size_t)sr0 * FDIM + c];
  }

  // per-lane gate biases (column-broadcast in C/D layout)
  const int colh = wave * 16 + (lane & 15);
  const float bsi = bih[colh]          + bhh[colh];
  const float bsf = bih[HDIM   + colh] + bhh[HDIM   + colh];
  const float bsg = bih[2*HDIM + colh] + bhh[2*HDIM + colh];
  const float bso = bih[3*HDIM + colh] + bhh[3*HDIM + colh];

  v8f c_st;
#pragma unroll
  for (int r = 0; r < 8; ++r) c_st[r] = 0.0f;

  __syncthreads();

  for (int t = 0; t < DEG; ++t) {
    const _Float16* sXTcur = (t & 1) ? sXT1 : sXT0;
    _Float16*       sXTnxt = (t & 1) ? sXT0 : sXT1;
    const bool havepre = (t + 1 < DEG);

    // ---- issue the t+1 gather now; it completes under the WMMA block ----
    float4 pf0, pf1;   // fp32 path: 2 float4 per thread (16 rows x 32 float4)
    uint4  pfh;        // f16 fallback path: 1x16B per thread (16 rows x 16 chunks)
    if (havepre) {
      if constexpr (sizeof(TI) == 4) {
        int i0 = tid, i1 = tid + 256;
        int m0 = i0 >> 5, c0 = i0 & 31;
        int m1 = i1 >> 5, c1 = i1 & 31;
        int s0 = src[(n0 + m0) * DEG + t + 1];
        int s1 = src[(n0 + m1) * DEG + t + 1];
        pf0 = ((const float4*)xin)[(size_t)s0 * (FDIM / 4) + c0];
        pf1 = ((const float4*)xin)[(size_t)s1 * (FDIM / 4) + c1];
      } else {
        int m = tid >> 4, ch = tid & 15;           // 16B chunk per thread
        int sr = src[(n0 + m) * DEG + t + 1];
        const TI* g = xin + (size_t)sr * FDIM + ch * 8;
        _Float16* l = sXTnxt + m * LDW + ch * 8;
#ifdef HAVE_ASYNC_LDS
        // CDNA5 async copy: global -> LDS, tracked by ASYNCcnt, no VGPR round-trip
        __builtin_amdgcn_global_load_async_to_lds_b128(
            (gv4i*)(void*)g, (lv4i*)(void*)l, 0, 0);
#else
        pfh = *(const uint4*)g;
#endif
      }
    }

    // ---- gates: 32 WMMAs per wave per step ----
    v8f ai, af, ag, ao;
#pragma unroll
    for (int r = 0; r < 8; ++r) { ai[r]=0.f; af[r]=0.f; ag[r]=0.f; ao[r]=0.f; }

#pragma unroll
    for (int kc = 0; kc < 4; ++kc) {
      int k0 = kc * 32;
      v16h aX = load_A_frag(sXTcur, k0, lane);
      v16h aH = load_A_frag(sH,     k0, lane);
      ai = wmma_f16(aX, load_B_frag(sWih,            wave*16, k0, lane), ai);
      ai = wmma_f16(aH, whhF[0][kc], ai);
      af = wmma_f16(aX, load_B_frag(sWih, HDIM     + wave*16, k0, lane), af);
      af = wmma_f16(aH, whhF[1][kc], af);
      ag = wmma_f16(aX, load_B_frag(sWih, 2*HDIM   + wave*16, k0, lane), ag);
      ag = wmma_f16(aH, whhF[2][kc], ag);
      ao = wmma_f16(aX, load_B_frag(sWih, 3*HDIM   + wave*16, k0, lane), ao);
      ao = wmma_f16(aH, whhF[3][kc], ao);
    }

    // ---- in-register LSTM pointwise (i,f,g,o) ----
    float hv[8];
#pragma unroll
    for (int r = 0; r < 8; ++r) {
      float gi = fsigmoid(ai[r] + bsi);
      float gf = fsigmoid(af[r] + bsf);
      float gg = ftanh   (ag[r] + bsg);
      float go = fsigmoid(ao[r] + bso);
      float cc = gf * c_st[r] + gi * gg;
      c_st[r]  = cc;
      hv[r]    = go * ftanh(cc);
    }

    __syncthreads();                     // all reads of sH / sXTcur complete
    {
      int col = wave * 16 + (lane & 15);
      int mb  = (lane >> 4) * 8;
#pragma unroll
      for (int r = 0; r < 8; ++r)
        sH[(mb + r) * LDW + col] = (_Float16)hv[r];
    }
    // ---- commit the prefetched x_{t+1} into the other buffer ----
    if (havepre) {
      if constexpr (sizeof(TI) == 4) {
        int i0 = tid, i1 = tid + 256;
        int m0 = i0 >> 5, c0 = (i0 & 31) * 4;
        int m1 = i1 >> 5, c1 = (i1 & 31) * 4;
        h4v h0; h0[0]=(_Float16)pf0.x; h0[1]=(_Float16)pf0.y; h0[2]=(_Float16)pf0.z; h0[3]=(_Float16)pf0.w;
        h4v h1; h1[0]=(_Float16)pf1.x; h1[1]=(_Float16)pf1.y; h1[2]=(_Float16)pf1.z; h1[3]=(_Float16)pf1.w;
        *(h4v*)(sXTnxt + m0 * LDW + c0) = h0;
        *(h4v*)(sXTnxt + m1 * LDW + c1) = h1;
      } else {
#ifdef HAVE_ASYNC_LDS
#if __has_builtin(__builtin_amdgcn_s_wait_asynccnt)
        __builtin_amdgcn_s_wait_asynccnt(0);
#else
        asm volatile("s_wait_asynccnt 0x0" ::: "memory");
#endif
#else
        int m = tid >> 4, ch = tid & 15;
        *(uint4*)(sXTnxt + m * LDW + ch * 8) = pfh;
#endif
      }
    }
    __syncthreads();                     // sH + sXTnxt ready for next step
  }

  // ---- out = relu(aggr @ lin_l^T + b_l + x_n @ lin_r^T); reuse sWih for lin weights ----
  {
    const float4* l4 = (const float4*)wl;
    const float4* r4 = (const float4*)wr;
    for (int idx = tid; idx < HDIM * (FDIM / 4); idx += 256) {
      int r = idx >> 5;
      int c = (idx & 31) * 4;
      float4 vl = l4[idx];
      float4 vr = r4[idx];
      h4v hl; hl[0]=(_Float16)vl.x; hl[1]=(_Float16)vl.y; hl[2]=(_Float16)vl.z; hl[3]=(_Float16)vl.w;
      h4v hr; hr[0]=(_Float16)vr.x; hr[1]=(_Float16)vr.y; hr[2]=(_Float16)vr.z; hr[3]=(_Float16)vr.w;
      *(h4v*)(sWih + r * LDW + c) = hl;
      *(h4v*)(sWih + (HDIM + r) * LDW + c) = hr;
    }
  }
  __syncthreads();

  v8f acc;
#pragma unroll
  for (int r = 0; r < 8; ++r) acc[r] = 0.0f;
#pragma unroll
  for (int kc = 0; kc < 4; ++kc) {
    int k0 = kc * 32;
    acc = wmma_f16(load_A_frag(sH,  k0, lane), load_B_frag(sWih,        wave*16, k0, lane), acc);
    acc = wmma_f16(load_A_frag(sXN, k0, lane), load_B_frag(sWih, HDIM + wave*16, k0, lane), acc);
  }
  {
    int col = wave * 16 + (lane & 15);
    int mb  = (lane >> 4) * 8;
    float bb = bl[col];
#pragma unroll
    for (int r = 0; r < 8; ++r) {
      float v = acc[r] + bb;
      v = v > 0.0f ? v : 0.0f;
      out[(size_t)(n0 + mb + r) * HDIM + col] = (_Float16)v;
    }
  }
}

// head: out[N,64] = in[N,128] @ lin_w^T + b ; 128 threads = 4 waves, one 16-col tile each
__global__ void head_kernel(const _Float16* __restrict__ in,
                            const float*    __restrict__ lw,
                            const float*    __restrict__ lb,
                            float*          __restrict__ out)
{
  __shared__ _Float16 sIn[TM * LDW];
  __shared__ _Float16 sW [OUTD * LDW];
  const int tid  = threadIdx.x;          // 0..127
  const int lane = tid & 31;
  const int wave = tid >> 5;             // 0..3
  const int n0   = blockIdx.x * TM;

  for (int idx = tid; idx < TM * HDIM; idx += 128) {
    int m = idx >> 7, c = idx & 127;
    sIn[m * LDW + c] = in[(size_t)(n0 + m) * HDIM + c];
  }
  for (int idx = tid; idx < OUTD * HDIM; idx += 128) {
    int r = idx >> 7, c = idx & 127;
    sW[r * LDW + c] = (_Float16)lw[idx];
  }
  __syncthreads();

  v8f acc;
#pragma unroll
  for (int r = 0; r < 8; ++r) acc[r] = 0.0f;
#pragma unroll
  for (int kc = 0; kc < 4; ++kc) {
    int k0 = kc * 32;
    acc = wmma_f16(load_A_frag(sIn, k0, lane), load_B_frag(sW, wave * 16, k0, lane), acc);
  }
  int col = wave * 16 + (lane & 15);
  int mb  = (lane >> 4) * 8;
  float bb = lb[col];
#pragma unroll
  for (int r = 0; r < 8; ++r)
    out[(size_t)(n0 + mb + r) * OUTD + col] = acc[r] + bb;
}

extern "C" void kernel_launch(void* const* d_in, const int* in_sizes, int n_in,
                              void* d_out, int out_size, void* d_ws, size_t ws_size,
                              hipStream_t stream) {
  (void)in_sizes; (void)n_in; (void)out_size; (void)ws_size;
  const float* x    = (const float*)d_in[0];
  const int*   edge = (const int*)d_in[1];   // [2, N*DEG]; src = first N*DEG entries
  const float* wih1 = (const float*)d_in[2];
  const float* whh1 = (const float*)d_in[3];
  const float* bih1 = (const float*)d_in[4];
  const float* bhh1 = (const float*)d_in[5];
  const float* wl1  = (const float*)d_in[6];
  const float* bl1  = (const float*)d_in[7];
  const float* wr1  = (const float*)d_in[8];
  const float* wih2 = (const float*)d_in[9];
  const float* whh2 = (const float*)d_in[10];
  const float* bih2 = (const float*)d_in[11];
  const float* bhh2 = (const float*)d_in[12];
  const float* wl2  = (const float*)d_in[13];
  const float* bl2  = (const float*)d_in[14];
  const float* wr2  = (const float*)d_in[15];
  const float* lw   = (const float*)d_in[16];
  const float* lbv  = (const float*)d_in[17];

  _Float16* out1 = (_Float16*)d_ws;                       // [N,128] f16
  _Float16* out2 = out1 + (size_t)N_NODES * HDIM;         // [N,128] f16

  const size_t lds_bytes = (size_t)(G4 * LDW + 4 * TM * LDW) * sizeof(_Float16); // ~153 KB -> 2 WGs/WGP
  (void)hipFuncSetAttribute(reinterpret_cast<const void*>(&sage_lstm_layer<float>),
                            hipFuncAttributeMaxDynamicSharedMemorySize, (int)lds_bytes);
  (void)hipFuncSetAttribute(reinterpret_cast<const void*>(&sage_lstm_layer<_Float16>),
                            hipFuncAttributeMaxDynamicSharedMemorySize, (int)lds_bytes);

  sage_lstm_layer<float><<<NTILES, 256, lds_bytes, stream>>>(
      x,    edge, wih1, whh1, bih1, bhh1, wl1, bl1, wr1, out1);
  sage_lstm_layer<_Float16><<<NTILES, 256, lds_bytes, stream>>>(
      out1, edge, wih2, whh2, bih2, bhh2, wl2, bl2, wr2, out2);
  head_kernel<<<NTILES, 128, 0, stream>>>(out2, lw, lbv, (float*)d_out);
}
